// LSTMModel_20925080666727
// MI455X (gfx1250) — matrix operand order repaired and checked by reading, not compiled, not run on hardware
//
#include <hip/hip_runtime.h>

typedef __attribute__((ext_vector_type(16))) __bf16 v16bf;
typedef __attribute__((ext_vector_type(8)))  __bf16 v8bf;
typedef __attribute__((ext_vector_type(8)))  float  v8f;

#define B_ 256
#define T_ 128
#define D_ 256
#define H_ 512
#define G_ 2048
#define O_ 128

__device__ __forceinline__ unsigned short f2bf(float f) {
  unsigned int u = __float_as_uint(f);
  u += 0x7FFFu + ((u >> 16) & 1u);          // round-to-nearest-even
  return (unsigned short)(u >> 16);
}
__device__ __forceinline__ float sigm(float x) { return 1.0f / (1.0f + __expf(-x)); }
__device__ __forceinline__ float tanh_fast(float x) { return 2.0f / (1.0f + __expf(-2.0f * x)) - 1.0f; }

// ---------------- prep kernels ----------------
__global__ void k_f2bf(const float* __restrict__ in, unsigned short* __restrict__ out, long n) {
  long i = (long)blockIdx.x * blockDim.x + threadIdx.x;
  long stride = (long)gridDim.x * blockDim.x;
  for (; i < n; i += stride) out[i] = f2bf(in[i]);
}

__global__ void k_zero32(unsigned int* __restrict__ p, long n) {
  long i = (long)blockIdx.x * blockDim.x + threadIdx.x;
  long stride = (long)gridDim.x * blockDim.x;
  for (; i < n; i += stride) p[i] = 0u;
}

// Wcat[g][k] = (k < H) ? W_hh[g][k] : W_ih[g][k-H]   (bf16, row-major, row length K)
__global__ void k_build_wcat(const float* __restrict__ Whh, const float* __restrict__ Wih,
                             unsigned short* __restrict__ out, int K, int Kx, long total) {
  long i = (long)blockIdx.x * blockDim.x + threadIdx.x;
  long stride = (long)gridDim.x * blockDim.x;
  for (; i < total; i += stride) {
    long g = i / K;
    int  k = (int)(i - g * K);
    float v = (k < H_) ? Whh[g * H_ + k] : Wih[g * (long)Kx + (k - H_)];
    out[i] = f2bf(v);
  }
}

__global__ void k_bsum(const float* __restrict__ a, const float* __restrict__ b,
                       float* __restrict__ o, int n) {
  int i = blockIdx.x * blockDim.x + threadIdx.x;
  if (i < n) o[i] = a[i] + b[i];
}

// ---------------- fragment helpers ----------------
// A fragment (contiguous k order, same as B): lane -> row M=lane%16,
// halves 0..15 = K[aoff..aoff+15] (aoff = 16*(lane/16))
__device__ __forceinline__ v16bf load_a_frag(const unsigned short* __restrict__ h_in,
                                             const unsigned short* __restrict__ x_t,
                                             long xstride, int am, int aoff, int kb) {
  const unsigned short* arow = (kb < H_)
      ? (h_in + (size_t)am * H_ + kb)
      : (x_t + (size_t)am * xstride + (kb - H_));
  v8bf lo = *(const v8bf*)(arow + aoff);
  v8bf hi = *(const v8bf*)(arow + 8 + aoff);
  return __builtin_shufflevector(lo, hi, 0,1,2,3,4,5,6,7,8,9,10,11,12,13,14,15);
}

struct BFrag { v16bf b0, b1, b2, b3; };

__device__ __forceinline__ BFrag load_b_frag(const unsigned short* __restrict__ Wcat,
                                             int K, size_t gstep, int bn, int boff, int kb) {
  const unsigned short* wb = Wcat + (size_t)bn * K + kb + boff;
  BFrag f;
  f.b0 = *(const v16bf*)(wb);
  f.b1 = *(const v16bf*)(wb + gstep);
  f.b2 = *(const v16bf*)(wb + 2 * gstep);
  f.b3 = *(const v16bf*)(wb + 3 * gstep);
  return f;
}

#define WMMA4(A, BF)                                                                                   \
  do {                                                                                                 \
    acc0 = __builtin_amdgcn_wmma_f32_16x16x32_bf16(false, (A), false, (BF).b0, (short)0, acc0, false, false); \
    acc1 = __builtin_amdgcn_wmma_f32_16x16x32_bf16(false, (A), false, (BF).b1, (short)0, acc1, false, false); \
    acc2 = __builtin_amdgcn_wmma_f32_16x16x32_bf16(false, (A), false, (BF).b2, (short)0, acc2, false, false); \
    acc3 = __builtin_amdgcn_wmma_f32_16x16x32_bf16(false, (A), false, (BF).b3, (short)0, acc3, false, false); \
  } while (0)

// ---------------- fused LSTM step ----------------
// gates = [h_t | x_t] @ Wcat^T + bsum.
// Block = 4 waves sharing one 16(M)x16(h) tile. Wave w (uniform via
// readfirstlane) accumulates interleaved K-chunks {w, w+4, w+8, ...} across all
// 4 gates; partials reduced through LDS; epilogue split 4 rows/wave.
// grid: 512 blocks (16 m-tiles * 32 h-tiles) x 128 threads.
__global__ void __launch_bounds__(128) lstm_step(
    const unsigned short* __restrict__ h_in,
    const unsigned short* __restrict__ x_t, long xstride,
    const unsigned short* __restrict__ Wcat, int K,
    const float* __restrict__ bsum,
    float* __restrict__ c_state,
    unsigned short* __restrict__ h_out,
    unsigned short* __restrict__ hseq, long seqstride)
{
  __shared__ float red[4][4][16][16];   // [gate][kpart][row][col] = 16 KB

  const int lane   = threadIdx.x & 31;
  const int wave_u = __builtin_amdgcn_readfirstlane(threadIdx.x >> 5);  // uniform wave id
  const int tile = blockIdx.x;
  const int m0 = (tile >> 5) * 16;
  const int h0 = (tile & 31) * 16;

  const int am   = m0 + (lane & 15);
  const int aoff = (lane < 16) ? 0 : 16;
  const int bn   = h0 + (lane & 15);
  const int boff = (lane < 16) ? 0 : 16;
  const size_t gstep = (size_t)512 * K;

  const int nkw = (K >> 5) >> 2;        // chunks per wave: 6 (K=768) or 8 (K=1024), even

  v8f acc0 = {}, acc1 = {}, acc2 = {}, acc3 = {};

  // ping-pong buffers A/B, unrolled by 2 with peeled tail (no cur=nxt copies)
  v16bf aA, aB;
  BFrag bA, bB;
  {
    const int kb = wave_u << 5;                       // chunk slot 0
    aA = load_a_frag(h_in, x_t, xstride, am, aoff, kb);
    bA = load_b_frag(Wcat, K, gstep, bn, boff, kb);
  }
  for (int i = 0; i + 2 < nkw; i += 2) {
    const int kb1 = (wave_u + 4 * (i + 1)) << 5;
    aB = load_a_frag(h_in, x_t, xstride, am, aoff, kb1);
    bB = load_b_frag(Wcat, K, gstep, bn, boff, kb1);
    WMMA4(aA, bA);
    const int kb2 = (wave_u + 4 * (i + 2)) << 5;
    aA = load_a_frag(h_in, x_t, xstride, am, aoff, kb2);
    bA = load_b_frag(Wcat, K, gstep, bn, boff, kb2);
    WMMA4(aB, bB);
  }
  {
    const int kbl = (wave_u + 4 * (nkw - 1)) << 5;    // last chunk slot
    aB = load_a_frag(h_in, x_t, xstride, am, aoff, kbl);
    bB = load_b_frag(Wcat, K, gstep, bn, boff, kbl);
    WMMA4(aA, bA);
    WMMA4(aB, bB);
  }

  // dump per-wave partials (C/D layout: VGPR r -> M = r + 8*(lane>=16), N = lane%16)
  {
    const int mm = (lane < 16) ? 0 : 8;
    const int nn = lane & 15;
#pragma unroll
    for (int r = 0; r < 8; ++r) {
      red[0][wave_u][mm + r][nn] = acc0[r];
      red[1][wave_u][mm + r][nn] = acc1[r];
      red[2][wave_u][mm + r][nn] = acc2[r];
      red[3][wave_u][mm + r][nn] = acc3[r];
    }
  }
  __syncthreads();

  // epilogue: each wave owns 4 rows (64 cells -> 2 cells/lane)
#pragma unroll
  for (int j = 0; j < 2; ++j) {
    const int cell = j * 32 + lane;            // 0..63
    const int mm = wave_u * 4 + (cell >> 4);
    const int cc = cell & 15;
    const float si = red[0][0][mm][cc] + red[0][1][mm][cc] + red[0][2][mm][cc] + red[0][3][mm][cc];
    const float sf = red[1][0][mm][cc] + red[1][1][mm][cc] + red[1][2][mm][cc] + red[1][3][mm][cc];
    const float sg = red[2][0][mm][cc] + red[2][1][mm][cc] + red[2][2][mm][cc] + red[2][3][mm][cc];
    const float so = red[3][0][mm][cc] + red[3][1][mm][cc] + red[3][2][mm][cc] + red[3][3][mm][cc];

    const int row  = m0 + mm;
    const int ccol = h0 + cc;
    const size_t idx = (size_t)row * H_ + ccol;
    const float iv = sigm(si + bsum[ccol]);
    const float fv = sigm(sf + bsum[H_ + ccol]);
    const float gv = tanh_fast(sg + bsum[2 * H_ + ccol]);
    const float ov = sigm(so + bsum[3 * H_ + ccol]);
    const float cn = fv * c_state[idx] + iv * gv;
    c_state[idx] = cn;
    const float hn = ov * tanh_fast(cn);
    const unsigned short hb = f2bf(hn);
    h_out[idx] = hb;
    if (hseq) hseq[(size_t)row * seqstride + ccol] = hb;
  }
}

// ---------------- final FC: out = h2 @ fc_w^T + fc_b ----------------
// M=256, N=128, K=512 -> 128 wave tiles; grid 32 x 128 threads.
__global__ void __launch_bounds__(128) fc_kernel(
    const unsigned short* __restrict__ h2,
    const unsigned short* __restrict__ fcw,
    const float* __restrict__ fc_b,
    float* __restrict__ out)
{
  const int lane = threadIdx.x & 31;
  const int wave = __builtin_amdgcn_readfirstlane(threadIdx.x >> 5);
  const int tile = blockIdx.x * 4 + wave;
  const int m0 = (tile >> 3) * 16;
  const int n0 = (tile & 7) * 16;

  const int am   = m0 + (lane & 15);
  const int aoff = (lane < 16) ? 0 : 16;
  const int bn   = n0 + (lane & 15);
  const int boff = (lane < 16) ? 0 : 16;

  v8f acc = {};
#pragma unroll 4
  for (int kc = 0; kc < H_ / 32; ++kc) {
    const int kb = kc << 5;
    const unsigned short* arow = h2 + (size_t)am * H_ + kb;
    v8bf alo = *(const v8bf*)(arow + aoff);
    v8bf ahi = *(const v8bf*)(arow + 8 + aoff);
    v16bf a = __builtin_shufflevector(alo, ahi, 0,1,2,3,4,5,6,7,8,9,10,11,12,13,14,15);
    v16bf b = *(const v16bf*)(fcw + (size_t)bn * H_ + kb + boff);
    acc = __builtin_amdgcn_wmma_f32_16x16x32_bf16(false, a, false, b, (short)0, acc, false, false);
  }

  const int col = n0 + (lane & 15);
  const float bias = fc_b[col];
  const int mbase = m0 + ((lane < 16) ? 0 : 8);
#pragma unroll
  for (int r = 0; r < 8; ++r)
    out[(size_t)(mbase + r) * O_ + col] = acc[r] + bias;
}

// ---------------- host ----------------
extern "C" void kernel_launch(void* const* d_in, const int* in_sizes, int n_in,
                              void* d_out, int out_size, void* d_ws, size_t ws_size,
                              hipStream_t stream)
{
  const float* x    = (const float*)d_in[0];
  const float* Wih0 = (const float*)d_in[1];
  const float* Whh0 = (const float*)d_in[2];
  const float* bih0 = (const float*)d_in[3];
  const float* bhh0 = (const float*)d_in[4];
  const float* Wih1 = (const float*)d_in[5];
  const float* Whh1 = (const float*)d_in[6];
  const float* bih1 = (const float*)d_in[7];
  const float* bhh1 = (const float*)d_in[8];
  const float* fcw  = (const float*)d_in[9];
  const float* fcb  = (const float*)d_in[10];
  float* out = (float*)d_out;

  size_t off = 0;
  auto alloc = [&](size_t bytes) -> void* {
    off = (off + 255) & ~(size_t)255;
    void* p = (char*)d_ws + off;
    off += bytes;
    return p;
  };

  unsigned short* xbf    = (unsigned short*)alloc((size_t)B_ * T_ * D_ * 2);      // 16 MB
  unsigned short* wcat0  = (unsigned short*)alloc((size_t)G_ * (H_ + D_) * 2);    // 3 MB
  unsigned short* wcat1  = (unsigned short*)alloc((size_t)G_ * (H_ + H_) * 2);    // 4 MB
  unsigned short* fcwbf  = (unsigned short*)alloc((size_t)O_ * H_ * 2);
  float*          bsum0  = (float*)alloc((size_t)G_ * 4);
  float*          bsum1  = (float*)alloc((size_t)G_ * 4);
  unsigned short* hping  = (unsigned short*)alloc((size_t)B_ * H_ * 2);
  unsigned short* hpong  = (unsigned short*)alloc((size_t)B_ * H_ * 2);
  float*          cstate = (float*)alloc((size_t)B_ * H_ * 4);
  unsigned short* h1seq  = (unsigned short*)alloc((size_t)B_ * T_ * H_ * 2);      // 33.5 MB

  // --- prep: bf16 conversions, fused weights, fused biases ---
  k_f2bf<<<4096, 256, 0, stream>>>(x, xbf, (long)B_ * T_ * D_);
  k_build_wcat<<<2048, 256, 0, stream>>>(Whh0, Wih0, wcat0, H_ + D_, D_, (long)G_ * (H_ + D_));
  k_build_wcat<<<2048, 256, 0, stream>>>(Whh1, Wih1, wcat1, H_ + H_, H_, (long)G_ * (H_ + H_));
  k_f2bf<<<256, 256, 0, stream>>>(fcw, fcwbf, (long)O_ * H_);
  k_bsum<<<G_ / 256, 256, 0, stream>>>(bih0, bhh0, bsum0, G_);
  k_bsum<<<G_ / 256, 256, 0, stream>>>(bih1, bhh1, bsum1, G_);

  // --- layer 0 ---
  k_zero32<<<256, 256, 0, stream>>>((unsigned int*)hping, (long)B_ * H_ * 2 / 4);
  k_zero32<<<512, 256, 0, stream>>>((unsigned int*)cstate, (long)B_ * H_);
  unsigned short* hin = hping;
  unsigned short* hout = hpong;
  for (int t = 0; t < T_; ++t) {
    lstm_step<<<512, 128, 0, stream>>>(
        hin, xbf + (size_t)t * D_, (long)T_ * D_,
        wcat0, H_ + D_, bsum0, cstate, hout,
        h1seq + (size_t)t * H_, (long)T_ * H_);
    unsigned short* tmp = hin; hin = hout; hout = tmp;
  }

  // --- layer 1 ---
  k_zero32<<<256, 256, 0, stream>>>((unsigned int*)hping, (long)B_ * H_ * 2 / 4);
  k_zero32<<<512, 256, 0, stream>>>((unsigned int*)cstate, (long)B_ * H_);
  hin = hping; hout = hpong;
  for (int t = 0; t < T_; ++t) {
    lstm_step<<<512, 128, 0, stream>>>(
        hin, h1seq + (size_t)t * H_, (long)T_ * H_,
        wcat1, H_ + H_, bsum1, cstate, hout,
        (unsigned short*)nullptr, 0);
    unsigned short* tmp = hin; hin = hout; hout = tmp;
  }

  // hin now holds h2[:, T-1, :]
  fc_kernel<<<32, 128, 0, stream>>>(hin, fcwbf, fcb, out);
}
